// Seq2SeqModel_64390149702179
// MI455X (gfx1250) — matrix operand order repaired and checked by reading, not compiled, not run on hardware
//
#include <hip/hip_runtime.h>
#include <cstdint>

// ---------------------------------------------------------------------------
// CDNA5 (gfx1250) seq2seq inference.
//  - bf16 WMMA (v_wmma_f32_16x16x32_bf16) for every GEMM; fp32 accumulation.
//  - B=16 == one WMMA tile row dim -> zero padding waste.
//  - attention GEMM hoisted out of the decode loop (c0 precompute).
//  - per-wave gate-column ownership keeps the LSTM nonlinearity in registers.
//  - async global->LDS staging (ASYNCcnt) for shared A operands, with
//    double-buffered prefetch of x_{t+1} in the encoder time loop.
// Workspace requirement: ~40 MB.
// ---------------------------------------------------------------------------

typedef __attribute__((ext_vector_type(16))) __bf16 v16bf;
typedef __attribute__((ext_vector_type(8)))  float  v8f;

#define DEV __device__ __forceinline__

constexpr int NB = 16;       // batch
constexpr int NS = 48;       // src length
constexpr int NT = 32;       // decode steps
constexpr float NEGINF = -1000000000.0f;

#if __has_builtin(__builtin_amdgcn_global_load_async_to_lds_b128)
#define HAVE_ASYNC 1
#else
#define HAVE_ASYNC 0
#endif

DEV __bf16 to_bf16(float x) {
  union { float f; unsigned u; } a; a.f = x;
  unsigned r = a.u + 0x7FFFu + ((a.u >> 16) & 1u);   // round-to-nearest-even
  union { unsigned short s; __bf16 b; } o; o.s = (unsigned short)(r >> 16);
  return o.b;
}
DEV float sigm(float x) { return 1.0f / (1.0f + __expf(-x)); }

DEV v8f wmma_bf16(v16bf a, v16bf b, v8f c) {
  return __builtin_amdgcn_wmma_f32_16x16x32_bf16(false, a, false, b,
                                                 (short)0, c, false, false);
}

// Issue a cooperative global->LDS copy of `bytes` (multiple of 16, both
// pointers 16B aligned). Async path uses GLOBAL_LOAD_ASYNC_TO_LDS_B128
// (tracked by ASYNCcnt); caller must lds_copy_wait() + barrier before use.
DEV void lds_copy_issue(const __bf16* g, __bf16* l, int bytes, int tid, int nth) {
  int chunks = bytes >> 4;
#if HAVE_ASYNC
  typedef int v4i __attribute__((vector_size(16)));
  typedef __attribute__((address_space(1))) v4i* gvp_t;   // global source
  typedef __attribute__((address_space(3))) v4i* lvp_t;   // LDS destination
  gvp_t gp = (gvp_t)g;
  lvp_t lp = (lvp_t)l;
  for (int i = tid; i < chunks; i += nth)
    __builtin_amdgcn_global_load_async_to_lds_b128(gp + i, lp + i, 0, 0);
#else
  const uint4* s = (const uint4*)g;
  uint4* d = (uint4*)l;
  for (int i = tid; i < chunks; i += nth) d[i] = s[i];
#endif
}

DEV void lds_copy_wait() {
#if HAVE_ASYNC
#if __has_builtin(__builtin_amdgcn_s_wait_asynccnt)
  __builtin_amdgcn_s_wait_asynccnt(0);
#else
  asm volatile("s_wait_asynccnt 0" ::: "memory");
#endif
#endif
}

// A fragment: 16x32 bf16 tile, row-major source at `base` (already offset to
// tile origin), leading dim `lda` (elements). ISA layout: lanes 0-15 -> M,
// kbase 0; lanes 16-31 -> same M rows, kbase 8; VGPR j<4: K=kbase+2j,
// VGPR j>=4: K=16+kbase+2(j-4). Pairs packed as dwords.
DEV v16bf load_a(const __bf16* base, int lda, int lane) {
  int m  = lane & 15;
  int kb = (lane >> 4) << 3;
  union { v16bf v; unsigned u[8]; } a;
  const unsigned* row = (const unsigned*)(base + (size_t)m * lda);
#pragma unroll
  for (int j = 0; j < 8; ++j) {
    int k = ((j >> 2) << 4) + kb + ((j & 3) << 1);
    a.u[j] = row[k >> 1];
  }
  return a.v;
}

// B fragment: 32x16 bf16 tile of a row-major [K x N] matrix (pre-transposed
// weights). Lane indexes N (lane & 15); element s indexes K in the same order
// as load_a: s<8 -> K=kbase+s, s>=8 -> K=16+kbase+(s-8), kbase = 8*(lane>>4).
DEV v16bf load_b(const __bf16* wt, int ldn, int lane) {
  int n  = lane & 15;
  int kb = (lane >> 4) << 3;
  v16bf b;
#pragma unroll
  for (int s = 0; s < 16; ++s) {
    int k = ((s >> 3) << 4) + kb + (s & 7);
    b[s] = wt[(size_t)k * ldn + n];
  }
  return b;
}

// --------------------------- weight prep -----------------------------------
// W [N,K] fp32 (torch layout, row stride ldw, col offset col0) -> Wt [K,N] bf16
__global__ void conv_tr_kernel(const float* __restrict__ W, __bf16* __restrict__ Wt,
                               int N, int K, int ldw, int col0) {
  int idx = blockIdx.x * blockDim.x + threadIdx.x;
  if (idx >= N * K) return;
  int k = idx / N, n = idx - k * N;                 // coalesced on output
  Wt[idx] = to_bf16(W[(size_t)n * ldw + col0 + k]);
}

__global__ void embed_enc_kernel(const int* __restrict__ src,
                                 const float* __restrict__ emb,
                                 __bf16* __restrict__ x) {
  int idx = blockIdx.x * blockDim.x + threadIdx.x;     // [S,B,256]
  if (idx >= NS * NB * 256) return;
  int e = idx & 255, sb = idx >> 8, b = sb & 15, s = sb >> 4;
  x[idx] = to_bf16(emb[(size_t)src[b * NS + s] * 256 + e]);
}

// --------------------------- encoder ---------------------------------------
struct EncDir {
  const __bf16 *wih_t, *whh_t;     // [din,1024], [256,1024] bf16
  const float  *bih, *bhh;
  float *hT, *cT;                  // [16,256]
  int ycol0, rev;
};

// grid = 2 (fwd/bwd), block = 512 (16 waves). Wave w owns hidden cols
// [16w,16w+16) and computes its 4 gate tiles; cell update stays in registers.
// x_t tiles are double-buffered in LDS with async prefetch of x_{t+1}.
__global__ __launch_bounds__(512) void enc_lstm_kernel(
    EncDir d0, EncDir d1, const __bf16* __restrict__ xin, int din,
    __bf16* __restrict__ ybf, float* __restrict__ yf32) {
  EncDir d = blockIdx.x ? d1 : d0;
  __shared__ float  sc[16 * 256];
  __shared__ __bf16 shb[16 * 256];
  __shared__ __align__(16) __bf16 sx[2][16 * 512];
  int tid = threadIdx.x, lane = tid & 31, w = tid >> 5;
  for (int i = tid; i < 16 * 256; i += 512) { sc[i] = 0.f; shb[i] = to_bf16(0.f); }
  int kx = din >> 5;
  int xbytes = 16 * din * 2;
  {
    int t0 = d.rev ? (NS - 1) : 0;
    lds_copy_issue(xin + (size_t)t0 * 16 * din, sx[0], xbytes, tid, 512);
    lds_copy_wait();
  }
  __syncthreads();
  for (int t = 0; t < NS; ++t) {
    int tt = d.rev ? (NS - 1 - t) : t;
    if (t + 1 < NS) {                      // prefetch x_{t+1} behind the WMMAs
      int tn = d.rev ? (NS - 2 - t) : (t + 1);
      lds_copy_issue(xin + (size_t)tn * 16 * din, sx[(t + 1) & 1], xbytes, tid, 512);
    }
    const __bf16* xr = sx[t & 1];
    v8f acc[4] = {};
    for (int k = 0; k < kx; ++k) {
      v16bf a = load_a(xr + k * 32, din, lane);
#pragma unroll
      for (int g = 0; g < 4; ++g) {
        v16bf b = load_b(d.wih_t + (size_t)(k * 32) * 1024 + g * 256 + w * 16, 1024, lane);
        acc[g] = wmma_bf16(a, b, acc[g]);
      }
    }
    for (int k = 0; k < 8; ++k) {
      v16bf a = load_a((const __bf16*)shb + k * 32, 256, lane);
#pragma unroll
      for (int g = 0; g < 4; ++g) {
        v16bf b = load_b(d.whh_t + (size_t)(k * 32) * 1024 + g * 256 + w * 16, 1024, lane);
        acc[g] = wmma_bf16(a, b, acc[g]);
      }
    }
    __syncthreads();                       // all shb / sx reads complete
    int nl = lane & 15, n = w * 16 + nl, hi = (lane >> 4) << 3;
#pragma unroll
    for (int r = 0; r < 8; ++r) {
      int m = r + hi;
      float gi = acc[0][r] + d.bih[n]       + d.bhh[n];
      float gf = acc[1][r] + d.bih[256 + n] + d.bhh[256 + n];
      float gg = acc[2][r] + d.bih[512 + n] + d.bhh[512 + n];
      float go = acc[3][r] + d.bih[768 + n] + d.bhh[768 + n];
      float cn = sigm(gf) * sc[m * 256 + n] + sigm(gi) * tanhf(gg);
      float hn = sigm(go) * tanhf(cn);
      sc[m * 256 + n]  = cn;
      shb[m * 256 + n] = to_bf16(hn);
      size_t yo = ((size_t)tt * 16 + m) * 512 + d.ycol0 + n;
      ybf[yo] = to_bf16(hn);
      if (yf32) yf32[yo] = hn;
      if (t == NS - 1) { d.hT[m * 256 + n] = hn; d.cT[m * 256 + n] = cn; }
    }
    lds_copy_wait();                       // x_{t+1} landed in LDS
    __syncthreads();
  }
}

// --------------------------- bridge ----------------------------------------
__global__ __launch_bounds__(512) void bridge_kernel(
    const float* hF, const float* cF, const float* hB, const float* cB,
    const __bf16* wbh_t, const float* bbh, const __bf16* wbc_t, const float* bbc,
    float* dec_h, float* dec_c, int* tok, const int* sos) {
  __shared__ __bf16 ah[16 * 512];
  __shared__ __bf16 ac[16 * 512];
  int tid = threadIdx.x, lane = tid & 31, w = tid >> 5;
  for (int i = tid; i < 16 * 512; i += 512) {
    int m = i >> 9, k = i & 511;
    ah[i] = to_bf16(k < 256 ? hF[m * 256 + k] : hB[m * 256 + k - 256]);
    ac[i] = to_bf16(k < 256 ? cF[m * 256 + k] : cB[m * 256 + k - 256]);
  }
  if (tid < 16) tok[tid] = *sos;
  __syncthreads();
  v8f a0 = {}, a1 = {};
  for (int k = 0; k < 16; ++k) {
    v16bf Ah = load_a((const __bf16*)ah + k * 32, 512, lane);
    v16bf Bh = load_b(wbh_t + (size_t)(k * 32) * 256 + w * 16, 256, lane);
    a0 = wmma_bf16(Ah, Bh, a0);
    v16bf Ac = load_a((const __bf16*)ac + k * 32, 512, lane);
    v16bf Bc = load_b(wbc_t + (size_t)(k * 32) * 256 + w * 16, 256, lane);
    a1 = wmma_bf16(Ac, Bc, a1);
  }
  int nl = lane & 15, n = w * 16 + nl, hi = (lane >> 4) << 3;
#pragma unroll
  for (int r = 0; r < 8; ++r) {
    int m = r + hi;
    float vh = a0[r] + bbh[n], vc = a1[r] + bbc[n];
    for (int l = 0; l < 4; ++l) {
      dec_h[(l * 16 + m) * 256 + n] = vh;
      dec_c[(l * 16 + m) * 256 + n] = vc;
    }
  }
}

// ------------------- attention constant: c0[b,s] ----------------------------
// c0 = sum_j (enc@Wa_e^T + ba)_j * enc_j   (loop-invariant over decode steps)
__global__ __launch_bounds__(256) void attn_c0_kernel(
    const __bf16* __restrict__ enc_bf, const float* __restrict__ enc_f,
    const __bf16* __restrict__ wae_t, const float* __restrict__ ba,
    float* __restrict__ c0) {
  __shared__ float prod[16 * 512];
  __shared__ __align__(16) __bf16 sAr[16 * 512];
  int tid = threadIdx.x, lane = tid & 31, w = tid >> 5;   // 8 waves
  const __bf16* arow = enc_bf + (size_t)blockIdx.x * 16 * 512;
  const float*  frow = enc_f  + (size_t)blockIdx.x * 16 * 512;
  lds_copy_issue(arow, sAr, 16 * 512 * 2, tid, 256);
  lds_copy_wait();
  __syncthreads();
  for (int tI = 0; tI < 4; ++tI) {
    int nt = w * 4 + tI;
    v8f acc = {};
    for (int k = 0; k < 16; ++k) {
      v16bf a = load_a((const __bf16*)sAr + k * 32, 512, lane);
      v16bf b = load_b(wae_t + (size_t)(k * 32) * 512 + nt * 16, 512, lane);
      acc = wmma_bf16(a, b, acc);
    }
    int nl = lane & 15, n = nt * 16 + nl, hi = (lane >> 4) << 3;
#pragma unroll
    for (int r = 0; r < 8; ++r) {
      int m = r + hi;
      prod[m * 512 + n] = (acc[r] + ba[n]) * frow[m * 512 + n];
    }
  }
  __syncthreads();
  if (tid < 16) {
    float s = 0.f;
    for (int j = 0; j < 512; ++j) s += prod[tid * 512 + j];
    c0[blockIdx.x * 16 + tid] = s;
  }
}

// --------------------------- per-step attention -----------------------------
__global__ __launch_bounds__(512) void attn_kernel(
    const int* __restrict__ src, const int* __restrict__ tok,
    const float* __restrict__ dec_emb, const float* __restrict__ dec_h,
    const __bf16* __restrict__ wah_t, const float* __restrict__ c0,
    const float* __restrict__ enc_f,
    __bf16* __restrict__ xin_bf, __bf16* __restrict__ login_bf,
    float* __restrict__ w_out, int step) {
  __shared__ __bf16 topb[16 * 256];
  __shared__ float  q[16 * 512];
  __shared__ float  sc[16 * NS];
  __shared__ float  wgt[16 * NS];
  int tid = threadIdx.x, lane = tid & 31, w = tid >> 5;
  const float* top = dec_h + 3 * 16 * 256;
  for (int i = tid; i < 16 * 256; i += 512) topb[i] = to_bf16(top[i]);
  for (int i = tid; i < 16 * 256; i += 512) {     // emb -> xin cols [0,256)
    int b = i >> 8, e = i & 255;
    xin_bf[b * 768 + e] = to_bf16(dec_emb[(size_t)tok[b] * 256 + e]);
  }
  __syncthreads();
  for (int tI = 0; tI < 2; ++tI) {                 // q = top @ Wa_h^T  [16,512]
    int nt = w * 2 + tI;
    v8f acc = {};
    for (int k = 0; k < 8; ++k) {
      v16bf a = load_a((const __bf16*)topb + k * 32, 256, lane);
      v16bf b = load_b(wah_t + (size_t)(k * 32) * 512 + nt * 16, 512, lane);
      acc = wmma_bf16(a, b, acc);
    }
    int nl = lane & 15, n = nt * 16 + nl, hi = (lane >> 4) << 3;
#pragma unroll
    for (int r = 0; r < 8; ++r) q[(r + hi) * 512 + n] = acc[r];
  }
  __syncthreads();
  for (int p = tid; p < 16 * NS; p += 512) {       // scores
    int b = p & 15, s = p >> 4;
    float v;
    if (src[b * NS + s] == 0) v = NEGINF;
    else {
      v = c0[s * 16 + b];
      const float* e  = enc_f + ((size_t)s * 16 + b) * 512;
      const float* qq = q + b * 512;
      float d = 0.f;
      for (int j = 0; j < 512; ++j) d += qq[j] * e[j];
      v += d;
    }
    sc[b * NS + s] = v;
  }
  __syncthreads();
  if (tid < 16) {                                  // masked softmax per row
    int b = tid;
    float mx = sc[b * NS];
    for (int s = 1; s < NS; ++s) mx = fmaxf(mx, sc[b * NS + s]);
    float sm = 0.f;
    for (int s = 0; s < NS; ++s) { float e = __expf(sc[b * NS + s] - mx); wgt[b * NS + s] = e; sm += e; }
    float inv = 1.f / sm;
    for (int s = 0; s < NS; ++s) {
      float ww = wgt[b * NS + s] * inv;
      wgt[b * NS + s] = ww;
      w_out[((size_t)b * NT + step) * NS + s] = ww;
    }
  }
  __syncthreads();
  {                                                // ctx = w @ enc_out
    int b = tid >> 5, cb = (tid & 31) << 4;
    for (int i = 0; i < 16; ++i) {
      int col = cb + i;
      float s = 0.f;
      for (int si = 0; si < NS; ++si)
        s += wgt[b * NS + si] * enc_f[((size_t)si * 16 + b) * 512 + col];
      __bf16 bf = to_bf16(s);
      xin_bf[b * 768 + 256 + col]   = bf;
      login_bf[b * 768 + 256 + col] = bf;
    }
  }
}

// --------------------------- decoder LSTM stack -----------------------------
struct DecLayer { const __bf16 *wih_t, *whh_t; const float *bih, *bhh; };

__global__ __launch_bounds__(512) void dec_lstm_kernel(
    DecLayer L0, DecLayer L1, DecLayer L2, DecLayer L3,
    const __bf16* __restrict__ xin0, float* __restrict__ dec_h,
    float* __restrict__ dec_c, __bf16* __restrict__ login_bf) {
  __shared__ __bf16 hb[16 * 256];
  __shared__ __bf16 xb[16 * 256];
  __shared__ __align__(16) __bf16 sx0[16 * 768];
  DecLayer Ls[4] = {L0, L1, L2, L3};
  int tid = threadIdx.x, lane = tid & 31, w = tid >> 5;
  lds_copy_issue(xin0, sx0, 16 * 768 * 2, tid, 512);   // overlap with hb fill
  for (int l = 0; l < 4; ++l) {
    DecLayer L = Ls[l];
    float* h = dec_h + l * 16 * 256;
    float* c = dec_c + l * 16 * 256;
    for (int i = tid; i < 16 * 256; i += 512) hb[i] = to_bf16(h[i]);
    if (l == 0) lds_copy_wait();
    __syncthreads();
    int din = (l == 0) ? 768 : 256;
    const __bf16* xa = (l == 0) ? (const __bf16*)sx0 : (const __bf16*)xb;
    v8f acc[4] = {};
    int kx = din >> 5;
    for (int k = 0; k < kx; ++k) {
      v16bf a = load_a(xa + k * 32, din, lane);
#pragma unroll
      for (int g = 0; g < 4; ++g) {
        v16bf b = load_b(L.wih_t + (size_t)(k * 32) * 1024 + g * 256 + w * 16, 1024, lane);
        acc[g] = wmma_bf16(a, b, acc[g]);
      }
    }
    for (int k = 0; k < 8; ++k) {
      v16bf a = load_a((const __bf16*)hb + k * 32, 256, lane);
#pragma unroll
      for (int g = 0; g < 4; ++g) {
        v16bf b = load_b(L.whh_t + (size_t)(k * 32) * 1024 + g * 256 + w * 16, 1024, lane);
        acc[g] = wmma_bf16(a, b, acc[g]);
      }
    }
    __syncthreads();
    int nl = lane & 15, n = w * 16 + nl, hi = (lane >> 4) << 3;
#pragma unroll
    for (int r = 0; r < 8; ++r) {
      int m = r + hi;
      float gi = acc[0][r] + L.bih[n]       + L.bhh[n];
      float gf = acc[1][r] + L.bih[256 + n] + L.bhh[256 + n];
      float gg = acc[2][r] + L.bih[512 + n] + L.bhh[512 + n];
      float go = acc[3][r] + L.bih[768 + n] + L.bhh[768 + n];
      float cn = sigm(gf) * c[m * 256 + n] + sigm(gi) * tanhf(gg);
      float hn = sigm(go) * tanhf(cn);
      c[m * 256 + n] = cn;
      h[m * 256 + n] = hn;
      xb[m * 256 + n] = to_bf16(hn);
      if (l == 3) login_bf[m * 768 + n] = to_bf16(hn);
    }
    __syncthreads();
  }
}

// --------------------------- logits + argmax --------------------------------
__global__ __launch_bounds__(256) void logits_kernel(
    const __bf16* __restrict__ login, const __bf16* __restrict__ wo_t,
    const float* __restrict__ bo, float* __restrict__ blkmax,
    int* __restrict__ blkarg) {
  __shared__ float v[16 * 128];
  __shared__ __align__(16) __bf16 sA[16 * 768];
  int tid = threadIdx.x, lane = tid & 31, w = tid >> 5;
  lds_copy_issue(login, sA, 16 * 768 * 2, tid, 256);
  lds_copy_wait();
  __syncthreads();
  int n0 = blockIdx.x * 128 + w * 16;
  v8f acc = {};
  for (int k = 0; k < 24; ++k) {
    v16bf a = load_a((const __bf16*)sA + k * 32, 768, lane);
    const __bf16* bp = wo_t + (size_t)(k * 32) * 16000 + n0;
    if (k < 23) __builtin_prefetch(bp + (size_t)32 * 16000, 0, 0);
    v16bf b = load_b(bp, 16000, lane);
    acc = wmma_bf16(a, b, acc);
  }
  int nl = lane & 15, hi = (lane >> 4) << 3;
#pragma unroll
  for (int r = 0; r < 8; ++r)
    v[(r + hi) * 128 + w * 16 + nl] = acc[r] + bo[n0 + nl];
  __syncthreads();
  if (tid < 16) {
    float best = v[tid * 128]; int bi = 0;
    for (int j = 1; j < 128; ++j) {
      float x = v[tid * 128 + j];
      if (x > best) { best = x; bi = j; }
    }
    blkmax[blockIdx.x * 16 + tid] = best;
    blkarg[blockIdx.x * 16 + tid] = blockIdx.x * 128 + bi;
  }
}

__global__ void argmax_kernel(const float* __restrict__ blkmax,
                              const int* __restrict__ blkarg,
                              int* __restrict__ tok, float* __restrict__ preds,
                              int step) {
  int b = threadIdx.x;
  if (b >= 16) return;
  float best = blkmax[b]; int bi = blkarg[b];
  for (int k = 1; k < 125; ++k) {                   // ascending cols: first max wins
    float x = blkmax[k * 16 + b];
    if (x > best) { best = x; bi = blkarg[k * 16 + b]; }
  }
  tok[b] = bi;
  preds[b * NT + step] = (float)bi;
}

// --------------------------- host driver ------------------------------------
extern "C" void kernel_launch(void* const* d_in, const int* in_sizes, int n_in,
                              void* d_out, int out_size, void* d_ws, size_t ws_size,
                              hipStream_t stream) {
  (void)in_sizes; (void)n_in; (void)out_size; (void)ws_size;
  const int*   src     = (const int*)d_in[0];
  const int*   sos     = (const int*)d_in[2];
  const float* enc_emb = (const float*)d_in[4];
  const float* dec_emb = (const float*)d_in[5];
  const float* eW[2][2][4];
  for (int l = 0; l < 2; ++l)
    for (int d = 0; d < 2; ++d)
      for (int j = 0; j < 4; ++j)
        eW[l][d][j] = (const float*)d_in[6 + (l * 2 + d) * 4 + j];
  const float* dWp[4][4];
  for (int l = 0; l < 4; ++l)
    for (int j = 0; j < 4; ++j) dWp[l][j] = (const float*)d_in[22 + l * 4 + j];
  const float* Wa  = (const float*)d_in[38]; const float* ba  = (const float*)d_in[39];
  const float* Wo  = (const float*)d_in[40]; const float* bo  = (const float*)d_in[41];
  const float* Wbh = (const float*)d_in[42]; const float* bbh = (const float*)d_in[43];
  const float* Wbc = (const float*)d_in[44]; const float* bbc = (const float*)d_in[45];

  // bump allocator over d_ws
  char* base = (char*)d_ws; size_t off = 0;
  auto take = [&](size_t bytes) -> void* {
    off = (off + 255) & ~(size_t)255;
    void* p = base + off; off += bytes; return p;
  };
  __bf16 *encWihT[2][2], *encWhhT[2][2];
  for (int l = 0; l < 2; ++l)
    for (int d = 0; d < 2; ++d) {
      encWihT[l][d] = (__bf16*)take((size_t)(l == 0 ? 256 : 512) * 1024 * 2);
      encWhhT[l][d] = (__bf16*)take((size_t)256 * 1024 * 2);
    }
  __bf16 *decWihT[4], *decWhhT[4];
  for (int l = 0; l < 4; ++l) {
    decWihT[l] = (__bf16*)take((size_t)(l == 0 ? 768 : 256) * 1024 * 2);
    decWhhT[l] = (__bf16*)take((size_t)256 * 1024 * 2);
  }
  __bf16* WaHT = (__bf16*)take((size_t)256 * 512 * 2);
  __bf16* WaET = (__bf16*)take((size_t)512 * 512 * 2);
  __bf16* WoT  = (__bf16*)take((size_t)768 * 16000 * 2);
  __bf16* WbhT = (__bf16*)take((size_t)512 * 256 * 2);
  __bf16* WbcT = (__bf16*)take((size_t)512 * 256 * 2);
  __bf16* embx = (__bf16*)take((size_t)NS * NB * 256 * 2);
  __bf16* y0   = (__bf16*)take((size_t)NS * NB * 512 * 2);
  __bf16* y1   = (__bf16*)take((size_t)NS * NB * 512 * 2);
  float*  yf   = (float*)take((size_t)NS * NB * 512 * 4);
  float* hF = (float*)take(16 * 256 * 4); float* cF = (float*)take(16 * 256 * 4);
  float* hB = (float*)take(16 * 256 * 4); float* cB = (float*)take(16 * 256 * 4);
  float* dec_h = (float*)take(4 * 16 * 256 * 4);
  float* dec_c = (float*)take(4 * 16 * 256 * 4);
  float* c0    = (float*)take(NS * NB * 4);
  int*   tok   = (int*)take(16 * 4);
  __bf16* xin   = (__bf16*)take(16 * 768 * 2);
  __bf16* login = (__bf16*)take(16 * 768 * 2);
  float* blkmax = (float*)take(125 * 16 * 4);
  int*   blkarg = (int*)take(125 * 16 * 4);

  auto conv = [&](const float* W, __bf16* Wt, int N, int K, int ldw, int col0) {
    int total = N * K;
    conv_tr_kernel<<<(total + 255) / 256, 256, 0, stream>>>(W, Wt, N, K, ldw, col0);
  };
  for (int l = 0; l < 2; ++l)
    for (int d = 0; d < 2; ++d) {
      int din = (l == 0) ? 256 : 512;
      conv(eW[l][d][0], encWihT[l][d], 1024, din, din, 0);
      conv(eW[l][d][1], encWhhT[l][d], 1024, 256, 256, 0);
    }
  for (int l = 0; l < 4; ++l) {
    int din = (l == 0) ? 768 : 256;
    conv(dWp[l][0], decWihT[l], 1024, din, din, 0);
    conv(dWp[l][1], decWhhT[l], 1024, 256, 256, 0);
  }
  conv(Wa,  WaHT, 512, 256, 768, 0);
  conv(Wa,  WaET, 512, 512, 768, 256);
  conv(Wo,  WoT, 16000, 768, 768, 0);
  conv(Wbh, WbhT, 256, 512, 512, 0);
  conv(Wbc, WbcT, 256, 512, 512, 0);

  {
    int total = NS * NB * 256;
    embed_enc_kernel<<<(total + 255) / 256, 256, 0, stream>>>(src, enc_emb, embx);
  }
  {
    EncDir f{encWihT[0][0], encWhhT[0][0], eW[0][0][2], eW[0][0][3], hF, cF, 0, 0};
    EncDir b{encWihT[0][1], encWhhT[0][1], eW[0][1][2], eW[0][1][3], hB, cB, 256, 1};
    enc_lstm_kernel<<<2, 512, 0, stream>>>(f, b, embx, 256, y0, nullptr);
  }
  {
    EncDir f{encWihT[1][0], encWhhT[1][0], eW[1][0][2], eW[1][0][3], hF, cF, 0, 0};
    EncDir b{encWihT[1][1], encWhhT[1][1], eW[1][1][2], eW[1][1][3], hB, cB, 256, 1};
    enc_lstm_kernel<<<2, 512, 0, stream>>>(f, b, y0, 512, y1, yf);
  }
  bridge_kernel<<<1, 512, 0, stream>>>(hF, cF, hB, cB, WbhT, bbh, WbcT, bbc,
                                       dec_h, dec_c, tok, sos);
  attn_c0_kernel<<<NS, 256, 0, stream>>>(y1, yf, WaET, ba, c0);

  DecLayer DL[4];
  for (int l = 0; l < 4; ++l)
    DL[l] = DecLayer{decWihT[l], decWhhT[l], dWp[l][2], dWp[l][3]};

  float* preds = (float*)d_out;              // [B,T] as float
  float* w_out = (float*)d_out + NB * NT;    // [B,T,S]
  for (int t = 0; t < NT; ++t) {
    attn_kernel<<<1, 512, 0, stream>>>(src, tok, dec_emb, dec_h, WaHT, c0, yf,
                                       xin, login, w_out, t);
    dec_lstm_kernel<<<1, 512, 0, stream>>>(DL[0], DL[1], DL[2], DL[3],
                                           xin, dec_h, dec_c, login);
    logits_kernel<<<125, 256, 0, stream>>>(login, WoT, bo, blkmax, blkarg);
    argmax_kernel<<<1, 32, 0, stream>>>(blkmax, blkarg, tok, preds, t);
  }
}
